// Easy_PCD_hidden_state_51049981281493
// MI455X (gfx1250) — compile-verified, matrix operand order checked by reading, not run on hardware
//
#include <hip/hip_runtime.h>

#define NF 32
#define DG 8

typedef __attribute__((ext_vector_type(16))) __bf16 v16bf;
typedef __attribute__((ext_vector_type(8)))  __bf16 v8bf;
typedef __attribute__((ext_vector_type(8)))  float  v8f;
typedef __attribute__((ext_vector_type(4)))  unsigned int v4u;
typedef __attribute__((ext_vector_type(8)))  int v8i;
typedef __attribute__((ext_vector_type(4)))  int v4i;

// ---------------------------------------------------------------------------
// One-time weight repack: (Cout, Cin, 3, 3) fp32 -> bf16 in WMMA B-fragment
// order:  idx = ((nt*ksteps + s)*32 + lane)*16 + e
// K ordering is tap-major: kg = tap*Cin + c  (tap = ky*3+kx).
// Zero-pads columns o >= Cout (e.g. 216 -> 224).
// ---------------------------------------------------------------------------
__global__ void __launch_bounds__(256)
repack_weights_kernel(const float* __restrict__ w, __bf16* __restrict__ wpk,
                      int Cout, int Cin)
{
  const int Ktot   = Cin * 9;
  const int ksteps = Ktot >> 5;
  const int ntiles = (Cout + 15) >> 4;
  const int total  = ntiles * ksteps * 512;
  const int i = blockIdx.x * 256 + threadIdx.x;
  if (i >= total) return;
  const int e    = i & 15;
  const int lane = (i >> 4) & 31;
  const int s    = (i >> 9) % ksteps;
  const int nt   = (i >> 9) / ksteps;
  const int hf   = lane >> 4;
  const int col  = lane & 15;
  const int kl   = e + 8 * hf + ((e & 8) ? 8 : 0);
  const int kg   = s * 32 + kl;
  const int tap  = kg / Cin;
  const int c    = kg - tap * Cin;
  const int o    = nt * 16 + col;
  const float v  = (o < Cout) ? w[(long)o * Ktot + c * 9 + tap] : 0.f;
  wpk[i] = (__bf16)v;
}

// ---------------------------------------------------------------------------
// 3x3 conv (pad 1) as implicit GEMM on WMMA bf16.
// Phase 1: 64 pixels x (9*CIN) im2col gather -> LDS bf16 (tap-major K).
// Phase 2: 8 waves cover 4 M-tiles x NTILES N-tiles; per step:
//          2x ds_load_b128 (A) + 2x global_load_b128 (packed B) + 1 WMMA.
// ---------------------------------------------------------------------------
template<int CIN, int NTILES>
__global__ void __launch_bounds__(256)
conv3x3_wmma_kernel(const float* __restrict__ in0,
                    const float* __restrict__ in1,   // concat half (CIN==64)
                    const __bf16* __restrict__ wpk,  // packed weights
                    const float* __restrict__ bias,
                    float* __restrict__ out,
                    int Cout, int Hin, int Win, int Ho, int Wo,
                    int stride, int do_lrelu)
{
  constexpr int KTOT   = CIN * 9;
  constexpr int KSTEPS = KTOT / 32;
  __shared__ __bf16 smem[64 * KTOT];

  const int tid   = threadIdx.x;
  const int HWo   = Ho * Wo;
  const int pbase = blockIdx.x * 64;

  __builtin_prefetch(wpk, 0, 1);       // gfx1250 global_prefetch_b8

  // ---- phase 1: im2col to LDS; thread owns fixed pixel pl = tid & 63 ----
  {
    const int pl = tid & 63;
    const int p  = pbase + pl;
    const int bb = p / HWo; const int rr = p - bb * HWo;
    const int oy = rr / Wo; const int ox = rr - oy * Wo;
    const long inPlane = (long)Hin * Win;
    const float* base0 = in0 + (long)bb * NF * inPlane;   // 32 ch per tensor
    const float* base1 = (CIN == 64) ? (in1 + (long)bb * NF * inPlane) : nullptr;
    const int iyb = oy * stride - 1;
    const int ixb = ox * stride - 1;
    #pragma unroll 4
    for (int it = 0; it < KTOT / 4; ++it) {
      const int kg  = it * 4 + (tid >> 6);
      const int tap = kg / CIN;                 // compile-time shift
      const int c   = kg & (CIN - 1);           // compile-time mask
      const int ky  = tap / 3;
      const int kx  = tap - 3 * ky;
      const int iy  = iyb + ky;
      const int ix  = ixb + kx;
      float v = 0.f;
      if (iy >= 0 && iy < Hin && ix >= 0 && ix < Win) {
        if (CIN == 64 && c >= NF)
          v = base1[(long)(c - NF) * inPlane + (long)iy * Win + ix];
        else
          v = base0[(long)c * inPlane + (long)iy * Win + ix];
      }
      smem[pl * KTOT + kg] = (__bf16)v;
    }
  }
  __syncthreads();

  // ---- phase 2: WMMA tiles ----
  const int wv = tid >> 5, lane = tid & 31, hf = lane >> 4, mrow = lane & 15;
  #pragma unroll
  for (int wt = wv; wt < 4 * NTILES; wt += 8) {
    const int mt = wt & 3;
    const int nt = wt >> 2;
    v8f acc = {};
    const __bf16* arow = &smem[(mt * 16 + mrow) * KTOT];
    const __bf16* brow = wpk + ((long)(nt * KSTEPS) * 32 + lane) * 16;
    #pragma unroll
    for (int s = 0; s < KSTEPS; ++s) {
      v16bf af, bf;
      const v8bf alo = *(const v8bf*)(arow + s * 32 + 8 * hf);
      const v8bf ahi = *(const v8bf*)(arow + s * 32 + 8 * hf + 16);
      const v8bf blo = *(const v8bf*)(brow + s * 512);
      const v8bf bhi = *(const v8bf*)(brow + s * 512 + 8);
      #pragma unroll
      for (int e = 0; e < 8; ++e) {
        af[e] = alo[e]; af[e + 8] = ahi[e];
        bf[e] = blo[e]; bf[e + 8] = bhi[e];
      }
      acc = __builtin_amdgcn_wmma_f32_16x16x32_bf16(false, af, false, bf,
                                                    (short)0, acc, false, false);
    }
    // C/D: lanes 0-15 -> N=lane, M=r ; lanes 16-31 -> N=lane-16, M=r+8
    const int oc = nt * 16 + mrow;
    if (oc < Cout) {
      const float bv = bias[oc];
      #pragma unroll
      for (int r = 0; r < 8; ++r) {
        const int pp = pbase + mt * 16 + r + 8 * hf;
        const int b2 = pp / HWo; const int rr = pp - b2 * HWo;
        const int y2 = rr / Wo;  const int x2 = rr - y2 * Wo;
        float v = acc[r] + bv;
        if (do_lrelu) v = (v >= 0.f) ? v : 0.1f * v;
        out[((long)b2 * Cout + oc) * HWo + (long)y2 * Wo + x2] = v;
      }
    }
  }
}

// ---------------------------------------------------------------------------
// Modulated deformable conv (DCNv2) given precomputed om.
// Packed weights (2*9*512 bf16 = 18432 B) are staged into LDS by the Tensor
// Data Mover (TENSOR_LOAD_TO_LDS, tracked by TENSORcnt), overlapped with the
// phase-1 deformable sampling, so phase 2 runs entirely out of LDS.
// ---------------------------------------------------------------------------
__global__ void __launch_bounds__(256)
dcn_wmma_kernel(const float* __restrict__ x,     // (B, 32, H, W)
                const float* __restrict__ om,    // (B, 216, H, W)
                const __bf16* __restrict__ wpk,  // packed (Cout=32, Cin=32)
                const float* __restrict__ bias,
                float* __restrict__ out,
                int H, int W, int do_lrelu)
{
  __shared__ __bf16 smem[64 * 288];              // 36 KB sampled A
  __shared__ __bf16 smemW[2 * 9 * 512];          // 18 KB packed B (TDM dest)
  const int tid   = threadIdx.x;
  const int HW    = H * W;
  const int pbase = blockIdx.x * 64;

  // ---- issue TDM copy of packed weights: global -> LDS (one wave) ----
  if (tid == 0) {
    const unsigned lds_off = (unsigned)(size_t)(void*)&smemW[0];
    const unsigned long long ga = (unsigned long long)(const void*)wpk;
    v4u g0;
    g0[0] = 1u;                                      // count=1, user descriptor
    g0[1] = lds_off;                                 // lds_addr (bytes)
    g0[2] = (unsigned)(ga & 0xffffffffu);            // global_addr[31:0]
    g0[3] = (unsigned)((ga >> 32) & 0x01ffffffu)     // global_addr[56:32]
          | (2u << 30);                              // type = 2 ("image")
    v8i g1 = {};
    g1[0] = (3 << 16);                               // data_size = 8 bytes
    g1[1] = (int)((2304u & 0xffffu) << 16);          // tensor_dim0[15:0] = 2304
    g1[2] = (1 << 16);                               // tensor_dim1 = 1
    g1[3] = (int)(2304u << 16);                      // tile_dim0 = 2304
    g1[4] = 0;                                       // tile_dim1/2 unused (1-D)
    g1[5] = 2304;                                    // tensor_dim0_stride
    g1[6] = 0;
    g1[7] = 0;
    v4i z = {};
#if __clang_major__ >= 23
    v8i z8 = {};
    __builtin_amdgcn_tensor_load_to_lds(g0, g1, z, z, z8, 0);
#else
    __builtin_amdgcn_tensor_load_to_lds(g0, g1, z, z, 0);
#endif
  }

  // ---- phase 1: deformable bilinear sampling -> LDS (overlaps the DMA) ----
  {
    const int pl = tid & 63;
    const int p  = pbase + pl;
    const int bb = p / HW; const int r2 = p - bb * HW;
    const int yy = r2 / W; const int xx = r2 - yy * W;
    const float* xb  = x + (long)bb * NF * HW;
    const long   ob0 = (long)bb * 216 * HW + (long)yy * W + xx;
    #pragma unroll 2
    for (int it = 0; it < 18; ++it) {            // 72 (g,k) pairs / 4 per iter
      const int gk   = it * 4 + (tid >> 6);
      const int g    = gk / 9;
      const int kidx = gk - g * 9;
      const long ob  = ob0 + (long)(g * 27) * HW;
      const float offy = om[ob + (long)kidx * HW];
      const float offx = om[ob + (long)(9 + kidx) * HW];
      const float mm   = om[ob + (long)(18 + kidx) * HW];
      const float mask = 1.f / (1.f + __expf(-mm));
      const float py = offy + (float)yy + (float)(kidx / 3 - 1);
      const float px = offx + (float)xx + (float)(kidx % 3 - 1);
      const float y0f = floorf(py), x0f = floorf(px);
      const int   y0  = (int)y0f,   x0  = (int)x0f;
      const float dy = py - y0f, dx = px - x0f;
      const float w00 = (1.f - dy) * (1.f - dx) * mask;
      const float w01 = (1.f - dy) * dx * mask;
      const float w10 = dy * (1.f - dx) * mask;
      const float w11 = dy * dx * mask;
      const bool y0ok = (y0 >= 0) && (y0 < H);
      const bool y1ok = (y0 + 1 >= 0) && (y0 + 1 < H);
      const bool x0ok = (x0 >= 0) && (x0 < W);
      const bool x1ok = (x0 + 1 >= 0) && (x0 + 1 < W);
      #pragma unroll
      for (int c = 0; c < 4; ++c) {
        const float* plane = xb + (long)(g * 4 + c) * HW;
        float v = 0.f;
        if (y0ok && x0ok) v += w00 * plane[y0 * W + x0];
        if (y0ok && x1ok) v += w01 * plane[y0 * W + x0 + 1];
        if (y1ok && x0ok) v += w10 * plane[(y0 + 1) * W + x0];
        if (y1ok && x1ok) v += w11 * plane[(y0 + 1) * W + x0 + 1];
        smem[pl * 288 + kidx * 32 + (g * 4 + c)] = (__bf16)v;  // tap-major K
      }
    }
  }
  if (tid == 0) __builtin_amdgcn_s_wait_tensorcnt(0);   // TDM done before barrier
  __syncthreads();

  // ---- phase 2: (64 x 288) x (288 x 32) WMMA bf16, all operands in LDS ----
  const int wv = tid >> 5, lane = tid & 31, hf = lane >> 4, mrow = lane & 15;
  const int mt = wv >> 1, nt = wv & 1;
  v8f acc = {};
  const __bf16* arow = &smem[(mt * 16 + mrow) * 288];
  const __bf16* brow = &smemW[((nt * 9) * 32 + lane) * 16];
  #pragma unroll
  for (int s = 0; s < 9; ++s) {
    v16bf af, bf;
    const v8bf alo = *(const v8bf*)(arow + s * 32 + 8 * hf);
    const v8bf ahi = *(const v8bf*)(arow + s * 32 + 8 * hf + 16);
    const v8bf blo = *(const v8bf*)(brow + s * 512);
    const v8bf bhi = *(const v8bf*)(brow + s * 512 + 8);
    #pragma unroll
    for (int e = 0; e < 8; ++e) {
      af[e] = alo[e]; af[e + 8] = ahi[e];
      bf[e] = blo[e]; bf[e + 8] = bhi[e];
    }
    acc = __builtin_amdgcn_wmma_f32_16x16x32_bf16(false, af, false, bf,
                                                  (short)0, acc, false, false);
  }
  const int oc = nt * 16 + mrow;
  const float bv = bias[oc];
  #pragma unroll
  for (int r = 0; r < 8; ++r) {
    const int p  = pbase + mt * 16 + r + 8 * hf;
    const int bb = p / HW; const int r2 = p - bb * HW;
    const int yy = r2 / W; const int xx = r2 - yy * W;
    float v = acc[r] + bv;
    if (do_lrelu) v = (v >= 0.f) ? v : 0.1f * v;
    out[((long)bb * NF + oc) * HW + (long)yy * W + xx] = v;
  }
}

// ---------------------------------------------------------------------------
// Bilinear x2 upsample (half-pixel centers, edge clamp).
// ---------------------------------------------------------------------------
__global__ void __launch_bounds__(256)
bilinear_up2_kernel(const float* __restrict__ in, float* __restrict__ out,
                    int Bn, int C, int Hi, int Wi)
{
  const int Ho = Hi * 2, Wo = Wi * 2;
  const long total = (long)Bn * C * Ho * Wo;
  const long i = (long)blockIdx.x * 256 + threadIdx.x;
  if (i >= total) return;
  const int x2 = (int)(i % Wo);
  long t = i / Wo;
  const int y2 = (int)(t % Ho); t /= Ho;
  const int c  = (int)(t % C);
  const int b  = (int)(t / C);
  const float sy = (y2 + 0.5f) * 0.5f - 0.5f;
  const float sx = (x2 + 0.5f) * 0.5f - 0.5f;
  const float y0f = floorf(sy), x0f = floorf(sx);
  const int y0 = (int)y0f, x0 = (int)x0f;
  const float dy = sy - y0f, dx = sx - x0f;
  const int y0c = min(max(y0, 0), Hi - 1), y1c = min(max(y0 + 1, 0), Hi - 1);
  const int x0c = min(max(x0, 0), Wi - 1), x1c = min(max(x0 + 1, 0), Wi - 1);
  const float* pl = in + ((long)b * C + c) * Hi * Wi;
  const float v = (1.f - dy) * (1.f - dx) * pl[y0c * Wi + x0c]
                + (1.f - dy) * dx         * pl[y0c * Wi + x1c]
                + dy * (1.f - dx)         * pl[y1c * Wi + x0c]
                + dy * dx                 * pl[y1c * Wi + x1c];
  out[((long)b * C + c) * (long)Ho * Wo + (long)y2 * Wo + x2] = v;
}

// ---------------------------------------------------------------------------
extern "C" void kernel_launch(void* const* d_in, const int* in_sizes, int n_in,
                              void* d_out, int out_size, void* d_ws, size_t ws_size,
                              hipStream_t stream)
{
  const float* L1_fea = (const float*)d_in[0];
  const float* L1_off = (const float*)d_in[1];
  const float* L2_off = (const float*)d_in[2];
  const float* L3_off = (const float*)d_in[3];
  const float* offs   = (const float*)d_in[4];
  const float* w_l2c1 = (const float*)d_in[5];  const float* b_l2c1 = (const float*)d_in[6];
  const float* w_l2c2 = (const float*)d_in[7];  const float* b_l2c2 = (const float*)d_in[8];
  const float* w_l3c1 = (const float*)d_in[9];  const float* b_l3c1 = (const float*)d_in[10];
  const float* w_l3c2 = (const float*)d_in[11]; const float* b_l3c2 = (const float*)d_in[12];
  const float* w_l2f  = (const float*)d_in[13]; const float* b_l2f  = (const float*)d_in[14];
  const float* w_l1f  = (const float*)d_in[15]; const float* b_l1f  = (const float*)d_in[16];
  const float* l3_omw = (const float*)d_in[17]; const float* l3_omb = (const float*)d_in[18];
  const float* l3_w   = (const float*)d_in[19]; const float* l3_b   = (const float*)d_in[20];
  const float* l2_omw = (const float*)d_in[21]; const float* l2_omb = (const float*)d_in[22];
  const float* l2_w   = (const float*)d_in[23]; const float* l2_b   = (const float*)d_in[24];
  const float* l1_omw = (const float*)d_in[25]; const float* l1_omb = (const float*)d_in[26];
  const float* l1_w   = (const float*)d_in[27]; const float* l1_b   = (const float*)d_in[28];
  const float* cs_omw = (const float*)d_in[29]; const float* cs_omb = (const float*)d_in[30];
  const float* cs_w   = (const float*)d_in[31]; const float* cs_b   = (const float*)d_in[32];
  float* outp = (float*)d_out;
  (void)in_sizes; (void)n_in; (void)out_size; (void)ws_size;

  const int B = 4, H1 = 192, W1 = 192, H2 = 96, W2 = 96, H3 = 48, W3 = 48;
  const size_t sz96  = (size_t)B * NF * H2 * W2;
  const size_t sz48  = (size_t)B * NF * H3 * W3;
  const size_t sz192 = (size_t)B * NF * H1 * W1;
  const size_t szom  = (size_t)B * 216 * H1 * W1;

  float* ws = (float*)d_ws;
  size_t woff = 0;
  auto take = [&](size_t n) { float* pp = ws + woff; woff += n; return pp; };
  float* om   = take(szom);    // offset/mask tensor, reused per DCN level
  float* bufA = take(sz96);    // L2a, later L3u
  float* bufB = take(sz96);    // L2,  later L2f2
  float* bufC = take(sz48);    // L3a, later L3f
  float* bufD = take(sz48);    // L3
  float* bufE = take(sz96);    // L2f
  float* bufG = take(sz192);   // L1f
  float* bufH = take(sz192);   // L2u
  float* bufI = take(sz192);   // fused L1f

  // packed-weight region (bf16), all sizes multiples of 16 elements
  const size_t PK_S  = 2 * 9 * 512;    // Cout=32,  Cin=32  -> 9216
  const size_t PK_C  = 2 * 18 * 512;   // Cout=32,  Cin=64  -> 18432
  const size_t PK_OM = 14 * 9 * 512;   // Cout=216, Cin=32  -> 64512
  __bf16* pk = (__bf16*)take((8 * PK_S + 2 * PK_C + 4 * PK_OM) / 2 + 16);
  size_t pko = 0;
  auto takepk = [&](size_t n) { __bf16* q = pk + pko; pko += n; return q; };
  __bf16* pk_l2c1 = takepk(PK_S); __bf16* pk_l2c2 = takepk(PK_S);
  __bf16* pk_l3c1 = takepk(PK_S); __bf16* pk_l3c2 = takepk(PK_S);
  __bf16* pk_l3w  = takepk(PK_S); __bf16* pk_l2w  = takepk(PK_S);
  __bf16* pk_l1w  = takepk(PK_S); __bf16* pk_csw  = takepk(PK_S);
  __bf16* pk_l2f  = takepk(PK_C); __bf16* pk_l1f  = takepk(PK_C);
  __bf16* pk_om3  = takepk(PK_OM); __bf16* pk_om2 = takepk(PK_OM);
  __bf16* pk_om1  = takepk(PK_OM); __bf16* pk_omc = takepk(PK_OM);

  dim3 blk(256);
  auto rgrid = [](size_t n) { return dim3((unsigned)((n + 255) / 256)); };

  // ---- repack all weights (one pass, tiny) ----
  repack_weights_kernel<<<rgrid(PK_S), blk, 0, stream>>>(w_l2c1, pk_l2c1, 32, 32);
  repack_weights_kernel<<<rgrid(PK_S), blk, 0, stream>>>(w_l2c2, pk_l2c2, 32, 32);
  repack_weights_kernel<<<rgrid(PK_S), blk, 0, stream>>>(w_l3c1, pk_l3c1, 32, 32);
  repack_weights_kernel<<<rgrid(PK_S), blk, 0, stream>>>(w_l3c2, pk_l3c2, 32, 32);
  repack_weights_kernel<<<rgrid(PK_S), blk, 0, stream>>>(l3_w,   pk_l3w,  32, 32);
  repack_weights_kernel<<<rgrid(PK_S), blk, 0, stream>>>(l2_w,   pk_l2w,  32, 32);
  repack_weights_kernel<<<rgrid(PK_S), blk, 0, stream>>>(l1_w,   pk_l1w,  32, 32);
  repack_weights_kernel<<<rgrid(PK_S), blk, 0, stream>>>(cs_w,   pk_csw,  32, 32);
  repack_weights_kernel<<<rgrid(PK_C), blk, 0, stream>>>(w_l2f,  pk_l2f,  32, 64);
  repack_weights_kernel<<<rgrid(PK_C), blk, 0, stream>>>(w_l1f,  pk_l1f,  32, 64);
  repack_weights_kernel<<<rgrid(PK_OM), blk, 0, stream>>>(l3_omw, pk_om3, 216, 32);
  repack_weights_kernel<<<rgrid(PK_OM), blk, 0, stream>>>(l2_omw, pk_om2, 216, 32);
  repack_weights_kernel<<<rgrid(PK_OM), blk, 0, stream>>>(l1_omw, pk_om1, 216, 32);
  repack_weights_kernel<<<rgrid(PK_OM), blk, 0, stream>>>(cs_omw, pk_omc, 216, 32);

  const int np96 = B * H2 * W2, np48 = B * H3 * W3, np192 = B * H1 * W1;
  auto g64 = [](int n) { return dim3((unsigned)(n / 64)); };

  // ---- pyramid build ----
  conv3x3_wmma_kernel<32, 2><<<g64(np96), blk, 0, stream>>>(L1_fea, nullptr, pk_l2c1, b_l2c1,
      bufA, 32, H1, W1, H2, W2, 2, 1);
  conv3x3_wmma_kernel<32, 2><<<g64(np96), blk, 0, stream>>>(bufA, nullptr, pk_l2c2, b_l2c2,
      bufB, 32, H2, W2, H2, W2, 1, 1);
  conv3x3_wmma_kernel<32, 2><<<g64(np48), blk, 0, stream>>>(bufB, nullptr, pk_l3c1, b_l3c1,
      bufC, 32, H2, W2, H3, W3, 2, 1);
  conv3x3_wmma_kernel<32, 2><<<g64(np48), blk, 0, stream>>>(bufC, nullptr, pk_l3c2, b_l3c2,
      bufD, 32, H3, W3, H3, W3, 1, 1);

  // ---- L3: om conv + DCN (+lrelu) ----
  conv3x3_wmma_kernel<32, 14><<<g64(np48), blk, 0, stream>>>(L3_off, nullptr, pk_om3, l3_omb,
      om, 216, H3, W3, H3, W3, 1, 0);
  dcn_wmma_kernel<<<g64(np48), blk, 0, stream>>>(bufD, om, pk_l3w, l3_b, bufC, H3, W3, 1);

  // ---- L2: om conv + DCN, fuse with upsampled L3f ----
  conv3x3_wmma_kernel<32, 14><<<g64(np96), blk, 0, stream>>>(L2_off, nullptr, pk_om2, l2_omb,
      om, 216, H2, W2, H2, W2, 1, 0);
  dcn_wmma_kernel<<<g64(np96), blk, 0, stream>>>(bufB, om, pk_l2w, l2_b, bufE, H2, W2, 0);
  bilinear_up2_kernel<<<rgrid(sz96), blk, 0, stream>>>(bufC, bufA, B, NF, H3, W3);
  conv3x3_wmma_kernel<64, 2><<<g64(np96), blk, 0, stream>>>(bufE, bufA, pk_l2f, b_l2f,
      bufB, 32, H2, W2, H2, W2, 1, 1);

  // ---- L1: om conv + DCN, fuse with upsampled L2f ----
  conv3x3_wmma_kernel<32, 14><<<g64(np192), blk, 0, stream>>>(L1_off, nullptr, pk_om1, l1_omb,
      om, 216, H1, W1, H1, W1, 1, 0);
  dcn_wmma_kernel<<<g64(np192), blk, 0, stream>>>(L1_fea, om, pk_l1w, l1_b, bufG, H1, W1, 0);
  bilinear_up2_kernel<<<rgrid(sz192), blk, 0, stream>>>(bufB, bufH, B, NF, H2, W2);
  conv3x3_wmma_kernel<64, 2><<<g64(np192), blk, 0, stream>>>(bufG, bufH, pk_l1f, b_l1f,
      bufI, 32, H1, W1, H1, W1, 1, 0);

  // ---- cascading DCN (+lrelu) -> output ----
  conv3x3_wmma_kernel<32, 14><<<g64(np192), blk, 0, stream>>>(offs, nullptr, pk_omc, cs_omb,
      om, 216, H1, W1, H1, W1, 1, 0);
  dcn_wmma_kernel<<<g64(np192), blk, 0, stream>>>(bufI, om, pk_csw, cs_b, outp, H1, W1, 1);
}
